// GCN_71906342469896
// MI455X (gfx1250) — compile-verified, matrix operand order checked by reading, not run on hardware
//
#include <hip/hip_runtime.h>
#include <hip/hip_bf16.h>

// ---------------------------------------------------------------------------
// GCN layer for MI455X (gfx1250, wave32, WMMA)
//   temp = features @ W_mlp^T + b        -> WMMA f16 (K=128)
//   x    = l2norm(concat(pref, temp))    -> fused into MLP kernel + pref kernel
//   xw   = x @ W_conv                    -> WMMA f16 (K=64)
//   out  = leaky_relu(scatter_add(xw[src] -> dst))
// ---------------------------------------------------------------------------

typedef _Float16 v16h __attribute__((ext_vector_type(16)));
typedef _Float16 v8h  __attribute__((ext_vector_type(8)));
typedef float    v8f  __attribute__((ext_vector_type(8)));

// ---------------------------------------------------------------------------
// Kernel 1: temp = features @ W_mlp^T + b, then L2-normalize rows, store f16.
// grid.x = NI/16 tiles of 16 rows; block = 128 threads = 4 waves (4 N-tiles).
// ---------------------------------------------------------------------------
__global__ __launch_bounds__(128)
void mlp_norm_kernel(const float* __restrict__ F,      // [NI,128]
                     const float* __restrict__ Wm,     // [64,128]
                     const float* __restrict__ bm,     // [64]
                     _Float16*    __restrict__ x16,    // [NU+NI,64] f16 out
                     int NUi)
{
    __shared__ float tile[16 * 65];                    // pitch 65: conflict-free

    const int lane = threadIdx.x & 31;
    const int wave = threadIdx.x >> 5;                 // N-tile 0..3
    const int mt   = blockIdx.x;                       // M-tile
    const int half = lane >> 4;                        // 0/1 (K-half select)
    const int l16  = lane & 15;
    const int mrow = mt * 16 + l16;                    // A row (per-lane)
    const int ncol = wave * 16 + l16;                  // B column == W_mlp row

    const float* __restrict__ arow = F  + (size_t)mrow * 128;
    const float* __restrict__ brow = Wm + (size_t)ncol * 128;

    v8f acc = {};
    #pragma unroll
    for (int kk = 0; kk < 4; ++kk) {                   // K = 128 = 4 x 32
        const int kb = kk * 32 + half * 8;
        v16h a, b;
        #pragma unroll
        for (int j = 0; j < 8; ++j) {
            a[j]     = (_Float16)arow[kb + j];
            a[j + 8] = (_Float16)arow[kb + 16 + j];
            b[j]     = (_Float16)brow[kb + j];
            b[j + 8] = (_Float16)brow[kb + 16 + j];
        }
        acc = __builtin_amdgcn_wmma_f32_16x16x32_f16(
                  false, a, false, b, (short)0, acc, false, false);
    }

    // D layout: VGPR v -> row (v + 8*half), col l16 (within this N-tile)
    const float bias = bm[ncol];
    #pragma unroll
    for (int v = 0; v < 8; ++v)
        tile[(v + 8 * half) * 65 + ncol] = acc[v] + bias;

    __syncthreads();

    // Fused row-wise L2 normalize: 8 threads per row (16 rows x 8 = 128).
    const int r = threadIdx.x >> 3;                    // row 0..15
    const int g = threadIdx.x & 7;                     // 8-thread group
    float s = 0.f;
    #pragma unroll
    for (int j = 0; j < 8; ++j) {
        const float v = tile[r * 65 + g * 8 + j];
        s += v * v;
    }
    s += __shfl_xor(s, 1);
    s += __shfl_xor(s, 2);
    s += __shfl_xor(s, 4);
    const float inv = 1.0f / fmaxf(sqrtf(s), 1e-12f);

    _Float16* __restrict__ orow = x16 + (size_t)(NUi + mt * 16 + r) * 64 + g * 8;
    #pragma unroll
    for (int j = 0; j < 8; ++j)
        orow[j] = (_Float16)(tile[r * 65 + g * 8 + j] * inv);
}

// ---------------------------------------------------------------------------
// Kernel 2: L2-normalize preference rows -> f16 rows [0, NU).
// One wave per row; lane handles elements (lane, lane+32).
// ---------------------------------------------------------------------------
__global__ __launch_bounds__(256)
void pref_norm_kernel(const float* __restrict__ P,    // [NU,64]
                      _Float16*    __restrict__ x16,  // [NU+NI,64]
                      int NUi)
{
    const int row  = blockIdx.x * 8 + (threadIdx.x >> 5);
    const int lane = threadIdx.x & 31;
    if (row >= NUi) return;

    const float* __restrict__ p = P + (size_t)row * 64;
    const float a = p[lane];
    const float b = p[lane + 32];
    float s = a * a + b * b;
    #pragma unroll
    for (int m = 16; m; m >>= 1) s += __shfl_xor(s, m);
    const float inv = 1.0f / fmaxf(sqrtf(s), 1e-12f);

    _Float16* __restrict__ o = x16 + (size_t)row * 64;
    o[lane]      = (_Float16)(a * inv);
    o[lane + 32] = (_Float16)(b * inv);
}

// ---------------------------------------------------------------------------
// Kernel 3: xw = x @ W_conv.  M = NU+NI, N = K = 64 (2 WMMAs per wave).
// grid.x = (NU+NI)/16; block = 128 threads = 4 waves (4 N-tiles).
// ---------------------------------------------------------------------------
__global__ __launch_bounds__(128)
void conv_kernel(const _Float16* __restrict__ x16,    // [M,64] f16
                 const float*    __restrict__ Wc,     // [64,64]
                 float*          __restrict__ xw)     // [M,64] f32
{
    const int lane = threadIdx.x & 31;
    const int wave = threadIdx.x >> 5;
    const int mt   = blockIdx.x;
    const int half = lane >> 4;
    const int l16  = lane & 15;
    const int mrow = mt * 16 + l16;
    const int ncol = wave * 16 + l16;

    v8f acc = {};
    #pragma unroll
    for (int kk = 0; kk < 2; ++kk) {                   // K = 64 = 2 x 32
        const int kb = kk * 32 + half * 8;
        const v8h a0 = *(const v8h*)(x16 + (size_t)mrow * 64 + kb);
        const v8h a1 = *(const v8h*)(x16 + (size_t)mrow * 64 + kb + 16);
        v16h a, b;
        #pragma unroll
        for (int j = 0; j < 8; ++j) {
            a[j]     = a0[j];
            a[j + 8] = a1[j];
            b[j]     = (_Float16)Wc[(size_t)(kb + j) * 64 + ncol];
            b[j + 8] = (_Float16)Wc[(size_t)(kb + 16 + j) * 64 + ncol];
        }
        acc = __builtin_amdgcn_wmma_f32_16x16x32_f16(
                  false, a, false, b, (short)0, acc, false, false);
    }

    float* __restrict__ out = xw + (size_t)(mt * 16) * 64 + ncol;
    #pragma unroll
    for (int v = 0; v < 8; ++v)
        out[(size_t)(v + 8 * half) * 64] = acc[v];
}

// ---------------------------------------------------------------------------
// Kernel 4: zero the output accumulator (graph-safe, no memset API).
// ---------------------------------------------------------------------------
__global__ __launch_bounds__(256)
void zero_kernel(float* __restrict__ p, int n)
{
    const int i = blockIdx.x * 256 + threadIdx.x;
    if (i < n) p[i] = 0.f;
}

// ---------------------------------------------------------------------------
// Kernel 5: edge scatter-add. 64 consecutive threads = one edge's 64 channels
// -> fully coalesced 256B gather and 256B atomic region per edge. xw and out
// (25.6 MB each) both fit in the 192 MB L2, so atomics mostly stay on-chip.
// ---------------------------------------------------------------------------
__global__ __launch_bounds__(256)
void scatter_kernel(const float* __restrict__ xw,
                    const int*   __restrict__ ei,     // [2,E]
                    float*       __restrict__ out,
                    int E)
{
    const int e = blockIdx.x * 4 + (threadIdx.x >> 6);
    if (e >= E) return;
    const int c   = threadIdx.x & 63;
    const int src = ei[e];
    const int dst = ei[E + e];
    atomicAdd(out + (size_t)dst * 64 + c, xw[(size_t)src * 64 + c]);
}

// ---------------------------------------------------------------------------
// Kernel 6: leaky_relu epilogue (in place, after all atomics are ordered).
// ---------------------------------------------------------------------------
__global__ __launch_bounds__(256)
void lrelu_kernel(float* __restrict__ p, int n)
{
    const int i = blockIdx.x * 256 + threadIdx.x;
    if (i < n) {
        const float v = p[i];
        p[i] = v > 0.f ? v : 0.01f * v;
    }
}

// ---------------------------------------------------------------------------
// Launcher.  Inputs (setup_inputs order):
//  0: id_embedding (UNUSED by reference)   4: b_mlp [64]
//  1: features  [NI,128]                   5: W_conv [64,64]
//  2: preference [NU,64]                   6: edge_index [2,E] int32
//  3: W_mlp [64,128]
// Workspace: x16 f16 [(NU+NI)*64] then xw f32 [(NU+NI)*64].
// ---------------------------------------------------------------------------
extern "C" void kernel_launch(void* const* d_in, const int* in_sizes, int n_in,
                              void* d_out, int out_size, void* d_ws, size_t ws_size,
                              hipStream_t stream)
{
    const float* features = (const float*)d_in[1];
    const float* pref     = (const float*)d_in[2];
    const float* Wm       = (const float*)d_in[3];
    const float* bm       = (const float*)d_in[4];
    const float* Wc       = (const float*)d_in[5];
    const int*   ei       = (const int*)d_in[6];

    const int NIi  = in_sizes[1] / 128;
    const int NUi  = in_sizes[2] / 64;
    const int Ei   = in_sizes[6] / 2;
    const int Ntot = NUi + NIi;

    _Float16* x16 = (_Float16*)d_ws;
    float*    xw  = (float*)((char*)d_ws + (size_t)Ntot * 64 * sizeof(_Float16));
    float*    out = (float*)d_out;

    mlp_norm_kernel<<<NIi / 16, 128, 0, stream>>>(features, Wm, bm, x16, NUi);
    pref_norm_kernel<<<(NUi + 7) / 8, 256, 0, stream>>>(pref, x16, NUi);
    conv_kernel<<<Ntot / 16, 128, 0, stream>>>(x16, Wc, xw);
    zero_kernel<<<(out_size + 255) / 256, 256, 0, stream>>>(out, out_size);
    scatter_kernel<<<(Ei + 3) / 4, 256, 0, stream>>>(xw, ei, out, Ei);
    lrelu_kernel<<<(out_size + 255) / 256, 256, 0, stream>>>(out, out_size);
}
